// SequenceTwoPlayerAgent_68418829025327
// MI455X (gfx1250) — compile-verified, matrix operand order checked by reading, not run on hardware
//
// MI455X (gfx1250) fused inference kernel for SequenceTwoPlayerAgent.
//
// Roofline: ~20 GFLOP of GEMM work vs ~106 MB of unavoidable HBM traffic
// (59 MB input + 46 MB output) => memory floor ~4.6 us at 23.3 TB/s. The
// whole network is fused into ONE kernel: one wave32 per 16-sample tile, all
// intermediates in LDS. Input is staged through a 5-row ring buffer so the
// workgroup needs only 61 KB LDS => 5 single-wave WGs per 320 KB WGP.
// All GEMMs use v_wmma_f32_16x16x32_f16 (f16 in, f32 accumulate).
//
// Data-movement design: the K-dimension of every GEMM is ordered so each
// lane's A-fragment (halves K = {b..b+7, b+16..b+23}, CDNA5 16-bit A layout)
// is exactly two aligned 16-byte LDS runs -> two ds_load_b128 per fragment:
//   conv1: input ring [s][slot][x][16ch] (9ch zero-padded), K' = pixel*16+c
//   conv2: act1 stored as [s][pos][32ch], K' = (ky*3+kx)*32+c
//   fc/heads: naturally contiguous.
// Weights are pre-packed (once) into matching f16 B-fragment order.

#include <hip/hip_runtime.h>

typedef __attribute__((ext_vector_type(16))) _Float16 v16h;
typedef __attribute__((ext_vector_type(8)))  _Float16 v8h;
typedef __attribute__((ext_vector_type(8)))  float    v8f;

// CDNA5 16-bit A-matrix 16x32 layout (ISA 7.12.2): lane<16 holds row M=lane,
// K = {0..7,16..23}; lane>=16 holds row M=lane-16, K = {8..15,24..31}.
__device__ __forceinline__ int akmap(int i, int lane) {
  int k = (i < 8) ? i : (i + 8);
  return k + ((lane & 16) ? 8 : 0);
}

__device__ __forceinline__ v8f wmma16(v16h a, v16h b, v8f c) {
  return __builtin_amdgcn_wmma_f32_16x16x32_f16(false, a, false, b, (short)0, c,
                                                false, false);
}

__device__ __forceinline__ v16h combine(v8h lo, v8h hi) {
  return __builtin_shufflevector(lo, hi, 0, 1, 2, 3, 4, 5, 6, 7,
                                 8, 9, 10, 11, 12, 13, 14, 15);
}

__device__ __forceinline__ v16h afrag_contig(const _Float16* row_ptr, int kt,
                                             int lane) {
  const _Float16* p = row_ptr + kt * 32 + ((lane & 16) ? 8 : 0);
  return combine(*(const v8h*)p, *(const v8h*)(p + 16));
}

__device__ __forceinline__ v16h bfrag(const _Float16* w, int t, int lane) {
  return *(const v16h*)(w + ((size_t)t * 32 + lane) * 16);
}

// ---------------------------------------------------------------------------
// Prep kernel: pack f32 weights into f16 WMMA B-fragments with optional K
// permutation. mode 0: identity (K zero-padded). mode 1: conv1, K'=pp*16+c,
// orig k = c*25+pp, valid iff c<9 && pp<25 (W is [N][225]). mode 2: conv2,
// K'=pos*32+c, orig k = c*9+pos (W is [N][288]).
// ---------------------------------------------------------------------------
__global__ void pack_b_kernel(const float* __restrict__ W, int N, int K,
                              int ktiles, int ntiles, int mode,
                              _Float16* __restrict__ out) {
  int tid = blockIdx.x * blockDim.x + threadIdx.x;
  int total = ktiles * ntiles * 32 * 16;
  if (tid >= total) return;
  int i    = tid & 15;
  int lane = (tid >> 4) & 31;
  int t    = tid >> 9;                 // t = ktile*ntiles + ntile
  int ntile = t % ntiles;
  int ktile = t / ntiles;
  int kp = ktile * 32 + akmap(i, lane);
  int n  = ntile * 16 + (lane & 15);
  float v = 0.0f;
  if (n < N) {
    if (mode == 0) {
      if (kp < K) v = W[(size_t)n * K + kp];
    } else if (mode == 1) {
      int pp = kp >> 4, c = kp & 15;
      if (pp < 25 && c < 9) v = W[(size_t)n * 225 + c * 25 + pp];
    } else {
      int pos = kp >> 5, c = kp & 31;
      v = W[(size_t)n * 288 + c * 9 + pos];
    }
  }
  out[((size_t)t * 32 + lane) * 16 + i] = (_Float16)v;
}

// ---------------------------------------------------------------------------
struct Params {
  const float* gin;    // [B][9][10][10]
  const float* gjack;  // [B][7]
  const float *bc1, *bc2, *bf1, *bf2;
  const float *br1, *br2, *br3, *bo1, *bo2, *bo3, *bh1, *bh2, *bh3;
  const _Float16 *wc1, *wc2, *wf1, *wf2;
  const _Float16 *wr1, *wr2, *wr3, *wo1, *wo2, *wo3, *wh1, *wh2, *wh3;
  float* out;          // [B][700]
};

// Stage one input row y (all 16 samples) into ring slot: [s][slot][x][16ch].
__device__ __forceinline__ void stage_row(const Params& P, _Float16* s_inp,
                                          int b0, int y, int slot, int lane) {
  for (int idx = lane; idx < 160; idx += 32) {      // idx = s*10 + x
    int s = idx / 10, x = idx % 10;
    const float* g = P.gin + (size_t)(b0 + s) * 900 + y * 10 + x;
    v8h lo, hi = {};
#pragma unroll
    for (int c = 0; c < 8; ++c) lo[c] = (_Float16)g[c * 100];
    hi[0] = (_Float16)g[800];
    _Float16* d = s_inp + (s * 50 + slot * 10 + x) * 16;
    *(v8h*)d       = lo;
    *(v8h*)(d + 8) = hi;
  }
}

// One head: 64->32->16->n3, no activations. f32 results into s_res[16][16].
__device__ void run_head(const _Float16* __restrict__ s_in,   // [16][64]
                         _Float16* s_t1, _Float16* s_t2,      // [16][32],[16][16]
                         float* s_res,                        // [16][16] f32
                         const _Float16* w1, const float* b1,
                         const _Float16* w2, const float* b2,
                         const _Float16* w3, const float* b3, int n3,
                         int lane) {
  const int mrow = lane & 15;
  const int msel = (lane >> 4) & 1;
  {
    v8f acc[2] = {};
#pragma unroll
    for (int kt = 0; kt < 2; ++kt) {
      v16h a = afrag_contig(s_in + mrow * 64, kt, lane);
#pragma unroll
      for (int nt = 0; nt < 2; ++nt)
        acc[nt] = wmma16(a, bfrag(w1, kt * 2 + nt, lane), acc[nt]);
    }
#pragma unroll
    for (int nt = 0; nt < 2; ++nt) {
      float bb = b1[nt * 16 + mrow];
#pragma unroll
      for (int j = 0; j < 8; ++j)
        s_t1[(j + 8 * msel) * 32 + nt * 16 + mrow] = (_Float16)(acc[nt][j] + bb);
    }
  }
  __syncthreads();
  {
    v16h a = afrag_contig(s_t1 + mrow * 32, 0, lane);
    v8f acc = {};
    acc = wmma16(a, bfrag(w2, 0, lane), acc);
    float bb = b2[mrow];
#pragma unroll
    for (int j = 0; j < 8; ++j)
      s_t2[(j + 8 * msel) * 16 + mrow] = (_Float16)(acc[j] + bb);
  }
  __syncthreads();
  {
    v8h lo = *(const v8h*)(s_t2 + mrow * 16 + ((lane & 16) ? 8 : 0));
    v8h zz = {};
    v8f acc = {};
    acc = wmma16(combine(lo, zz), bfrag(w3, 0, lane), acc);
    float bb = (mrow < n3) ? b3[mrow] : 0.0f;
#pragma unroll
    for (int j = 0; j < 8; ++j)
      s_res[(j + 8 * msel) * 16 + mrow] = acc[j] + bb;
  }
  __syncthreads();
}

__global__ __launch_bounds__(32, 1) void fused_agent(Params P) {
  extern __shared__ char smem[];
  // Live regions:
  _Float16* s_inp  = (_Float16*)(smem);          // ring [16][5][10][16] 25600 B
  _Float16* s_act1 = (_Float16*)(smem + 25600);  // [16][36 pos][32ch] 36864 B
  // Ring region reuse (input dead after conv1):
  _Float16* s_pool = (_Float16*)(smem);          // [16][256]
  _Float16* s_cat  = (_Float16*)(smem + 8192);   // [16][160]
  _Float16* s_fc2  = (_Float16*)(smem + 13312);  // [16][64]
  _Float16* s_t1   = (_Float16*)(smem + 15360);  // [16][32]
  _Float16* s_t2   = (_Float16*)(smem + 16384);  // [16][16]
  float*    s_row  = (float*)(smem + 16896);     // [16][16]
  float*    s_col  = (float*)(smem + 17920);     // [16][16]
  float*    s_hand = (float*)(smem + 18944);     // [16][16]
  float*    s_hr   = (float*)(smem + 19968);     // [16][70] hand x row

  const int lane = threadIdx.x;
  const int mrow = lane & 15;
  const int msel = (lane >> 4) & 1;
  const int hsel = (lane & 16) ? 8 : 0;
  const int b0   = blockIdx.x * 16;

  // --- prime ring with input rows 0..3 -------------------------------------
#pragma unroll 1
  for (int y = 0; y < 4; ++y) stage_row(P, s_inp, b0, y, y, lane);
  __syncthreads();

  // --- conv1 (+ReLU): per output row y0, GEMM M=16 N=32 K=416 (=26 pix*16c)
  // ktile t covers window pixels 2t, 2t+1 (pixel 25 = pad: zero weights,
  // clamped finite read). 12 accumulators (6 x-positions x 2 ntiles).
#pragma unroll 1
  for (int y0 = 0; y0 < 6; ++y0) {
    {   // stage input row y0+4 into ring slot (y0+4)%5
      int y = y0 + 4;
      stage_row(P, s_inp, b0, y, (y < 5) ? y : y - 5, lane);
    }
    __syncthreads();
    v8f acc[6][2] = {};
    for (int kt = 0; kt < 13; ++kt) {
      v16h bf0 = bfrag(P.wc1, kt * 2 + 0, lane);
      v16h bf1 = bfrag(P.wc1, kt * 2 + 1, lane);
      int p0 = 2 * kt, p1 = (2 * kt + 1 < 25) ? 2 * kt + 1 : 24;  // window pix
      int py0 = y0 + p0 / 5, py1 = y0 + p1 / 5;
      int sl0 = (py0 < 5) ? py0 : py0 - 5;       // ring slots
      int sl1 = (py1 < 5) ? py1 : py1 - 5;
#pragma unroll
      for (int x0 = 0; x0 < 6; ++x0) {
        const _Float16* rowp = s_inp + mrow * 800;
        const _Float16* c0 = rowp + (sl0 * 10 + x0 + p0 % 5) * 16 + hsel;
        const _Float16* c1 = rowp + (sl1 * 10 + x0 + p1 % 5) * 16 + hsel;
        v16h a = combine(*(const v8h*)c0, *(const v8h*)c1);
        acc[x0][0] = wmma16(a, bf0, acc[x0][0]);
        acc[x0][1] = wmma16(a, bf1, acc[x0][1]);
      }
    }
#pragma unroll
    for (int x0 = 0; x0 < 6; ++x0)
#pragma unroll
      for (int nt = 0; nt < 2; ++nt) {
        float bias = P.bc1[nt * 16 + mrow];
#pragma unroll
        for (int j = 0; j < 8; ++j) {
          float v = acc[x0][nt][j] + bias;
          v = v > 0.0f ? v : 0.0f;
          s_act1[(j + 8 * msel) * 1152 + (y0 * 6 + x0) * 32 + nt * 16 + mrow] =
              (_Float16)v;
        }
      }
    __syncthreads();   // ring slot of row y0 is recycled next iteration
  }

  // --- conv2 (+ReLU) fused with 2x2 maxpool --------------------------------
  // GEMM M=16 N=64 K=288; ktile = spatial tap, 32 contiguous channels.
  // ntiles processed in pairs so one A-fragment feeds two WMMAs.
#pragma unroll 1
  for (int nh = 0; nh < 2; ++nh) {
    v16h bf[2][9];
#pragma unroll
    for (int n2 = 0; n2 < 2; ++n2)
#pragma unroll
      for (int kt = 0; kt < 9; ++kt)
        bf[n2][kt] = bfrag(P.wc2, kt * 4 + nh * 2 + n2, lane);
    const float bias0 = P.bc2[(nh * 2 + 0) * 16 + mrow];
    const float bias1 = P.bc2[(nh * 2 + 1) * 16 + mrow];
#pragma unroll 1
    for (int q = 0; q < 4; ++q) {
      const int qy = q >> 1, qx = q & 1;
      v8f pm0 = {}, pm1 = {};             // 0-init => ReLU folded into max
#pragma unroll
      for (int d = 0; d < 4; ++d) {
        const int py = 2 * qy + (d >> 1), px = 2 * qx + (d & 1);
        const _Float16* arow = s_act1 + mrow * 1152;
        v8f a0 = {}, a1 = {};
#pragma unroll
        for (int kt = 0; kt < 9; ++kt) {
          int pos = (py + kt / 3) * 6 + px + kt % 3;
          const _Float16* p = arow + pos * 32 + hsel;
          v16h a = combine(*(const v8h*)p, *(const v8h*)(p + 16));
          a0 = wmma16(a, bf[0][kt], a0);
          a1 = wmma16(a, bf[1][kt], a1);
        }
#pragma unroll
        for (int j = 0; j < 8; ++j) {
          float v0 = a0[j] + bias0, v1 = a1[j] + bias1;
          pm0[j] = v0 > pm0[j] ? v0 : pm0[j];
          pm1[j] = v1 > pm1[j] ? v1 : pm1[j];
        }
      }
#pragma unroll
      for (int j = 0; j < 8; ++j) {       // pooled flatten order: ch*4 + q
        int ch0 = (nh * 2 + 0) * 16 + mrow;
        int ch1 = (nh * 2 + 1) * 16 + mrow;
        s_pool[(j + 8 * msel) * 256 + ch0 * 4 + q] = (_Float16)pm0[j];
        s_pool[(j + 8 * msel) * 256 + ch1 * 4 + q] = (_Float16)pm1[j];
      }
    }
  }
  __syncthreads();

  // --- fc1: M=16 N=128 K=256, ReLU; A hoisted across 8 ntiles --------------
  {
    v8f acc[8] = {};
    for (int kt = 0; kt < 8; ++kt) {
      v16h a = afrag_contig(s_pool + mrow * 256, kt, lane);
#pragma unroll
      for (int nt = 0; nt < 8; ++nt)
        acc[nt] = wmma16(a, bfrag(P.wf1, kt * 8 + nt, lane), acc[nt]);
    }
#pragma unroll
    for (int nt = 0; nt < 8; ++nt) {
      float bias = P.bf1[nt * 16 + mrow];
#pragma unroll
      for (int j = 0; j < 8; ++j) {
        float v = acc[nt][j] + bias;
        v = v > 0.0f ? v : 0.0f;
        s_cat[(j + 8 * msel) * 160 + nt * 16 + mrow] = (_Float16)v;
      }
    }
  }
  // concat jack features into s_cat[128..134], zero-pad 135..159
  for (int idx = lane; idx < 16 * 32; idx += 32) {
    int s = idx >> 5, k = 128 + (idx & 31);
    float v = (k < 135) ? P.gjack[(size_t)(b0 + s) * 7 + (k - 128)] : 0.0f;
    s_cat[s * 160 + k] = (_Float16)v;
  }
  __syncthreads();

  // --- fc2: M=16 N=64 K=135->160, ReLU -------------------------------------
  {
    v8f acc[4] = {};
    for (int kt = 0; kt < 5; ++kt) {
      v16h a = afrag_contig(s_cat + mrow * 160, kt, lane);
#pragma unroll
      for (int nt = 0; nt < 4; ++nt)
        acc[nt] = wmma16(a, bfrag(P.wf2, kt * 4 + nt, lane), acc[nt]);
    }
#pragma unroll
    for (int nt = 0; nt < 4; ++nt) {
      float bias = P.bf2[nt * 16 + mrow];
#pragma unroll
      for (int j = 0; j < 8; ++j) {
        float v = acc[nt][j] + bias;
        v = v > 0.0f ? v : 0.0f;
        s_fc2[(j + 8 * msel) * 64 + nt * 16 + mrow] = (_Float16)v;
      }
    }
  }
  __syncthreads();

  // --- three heads (no activations) ----------------------------------------
  run_head(s_fc2, s_t1, s_t2, s_row,  P.wr1, P.br1, P.wr2, P.br2, P.wr3, P.br3, 10, lane);
  run_head(s_fc2, s_t1, s_t2, s_col,  P.wo1, P.bo1, P.wo2, P.bo2, P.wo3, P.bo3, 10, lane);
  run_head(s_fc2, s_t1, s_t2, s_hand, P.wh1, P.bh1, P.wh2, P.bh2, P.wh3, P.bh3,  7, lane);

  // --- einsum bh,br,bc->b(hrc) ---------------------------------------------
  // Stage 1: hr[s][h*10+r] = hand*row (70 per sample).
  for (int idx = lane; idx < 16 * 70; idx += 32) {
    int s = idx / 70, u = idx % 70;
    s_hr[idx] = s_hand[s * 16 + u / 10] * s_row[s * 16 + u % 10];
  }
  __syncthreads();
  // Stage 2: out = hr * col; chunk-index outer so the u/10,u%10 division is
  // computed once per 32-element chunk and reused across all 16 samples.
#pragma unroll 1
  for (int t = 0; t < 22; ++t) {
    int u = t * 32 + lane;
    int hr = u / 10, c = u % 10;
    bool ok = u < 700;
#pragma unroll 1
    for (int s = 0; s < 16; ++s) {
      if (ok)
        P.out[(size_t)(b0 + s) * 700 + u] = s_hr[s * 70 + hr] * s_col[s * 16 + c];
    }
  }
}

// ---------------------------------------------------------------------------
extern "C" void kernel_launch(void* const* d_in, const int* in_sizes, int n_in,
                              void* d_out, int out_size, void* d_ws, size_t ws_size,
                              hipStream_t stream) {
  _Float16* ws = (_Float16*)d_ws;

  struct PackDesc { int in_idx, N, K, kt, nt, mode; size_t off; };
  const PackDesc packs[13] = {
    { 2,  32, 416, 13, 2, 1,     0},   // conv1_w  K' = pixel*16 + ch
    { 4,  64, 288,  9, 4, 2, 13312},   // conv2_w  K' = tap*32 + ch
    { 6, 128, 256,  8, 8, 0, 31744},   // fc1_w
    { 8,  64, 135,  5, 4, 0, 64512},   // fc2_w (K zero-padded to 160)
    {10,  32,  64,  2, 2, 0, 74752},   // row1_w
    {12,  16,  32,  1, 1, 0, 76800},   // row2_w
    {14,  10,  16,  1, 1, 0, 77312},   // row3_w (N pad 16, K pad 32)
    {16,  32,  64,  2, 2, 0, 77824},   // col1_w
    {18,  16,  32,  1, 1, 0, 79872},   // col2_w
    {20,  10,  16,  1, 1, 0, 80384},   // col3_w
    {22,  32,  64,  2, 2, 0, 80896},   // hand1_w
    {24,  16,  32,  1, 1, 0, 82944},   // hand2_w
    {26,   7,  16,  1, 1, 0, 83456},   // hand3_w
  };
  for (int m = 0; m < 13; ++m) {
    const PackDesc& p = packs[m];
    int total = p.kt * p.nt * 512;
    pack_b_kernel<<<(total + 255) / 256, 256, 0, stream>>>(
        (const float*)d_in[p.in_idx], p.N, p.K, p.kt, p.nt, p.mode, ws + p.off);
  }

  Params P;
  P.gin = (const float*)d_in[0];
  P.gjack = (const float*)d_in[1];
  P.bc1 = (const float*)d_in[3];  P.bc2 = (const float*)d_in[5];
  P.bf1 = (const float*)d_in[7];  P.bf2 = (const float*)d_in[9];
  P.br1 = (const float*)d_in[11]; P.br2 = (const float*)d_in[13]; P.br3 = (const float*)d_in[15];
  P.bo1 = (const float*)d_in[17]; P.bo2 = (const float*)d_in[19]; P.bo3 = (const float*)d_in[21];
  P.bh1 = (const float*)d_in[23]; P.bh2 = (const float*)d_in[25]; P.bh3 = (const float*)d_in[27];
  P.wc1 = ws + packs[0].off;  P.wc2 = ws + packs[1].off;
  P.wf1 = ws + packs[2].off;  P.wf2 = ws + packs[3].off;
  P.wr1 = ws + packs[4].off;  P.wr2 = ws + packs[5].off;  P.wr3 = ws + packs[6].off;
  P.wo1 = ws + packs[7].off;  P.wo2 = ws + packs[8].off;  P.wo3 = ws + packs[9].off;
  P.wh1 = ws + packs[10].off; P.wh2 = ws + packs[11].off; P.wh3 = ws + packs[12].off;
  P.out = (float*)d_out;

  int B = in_sizes[0] / 900;           // 16384
  int nblk = B / 16;                   // 1024 waves, one 16-sample tile each
  size_t lds_bytes = 62464;            // 61 KB -> 5 workgroups per WGP
  fused_agent<<<nblk, 32, lds_bytes, stream>>>(P);
}
